// InferenceQuantLinear_83081847373963
// MI455X (gfx1250) — compile-verified, matrix operand order checked by reading, not run on hardware
//
#include <hip/hip_runtime.h>

// ---------------------------------------------------------------------------
// Quantized (int4) linear layer as an f16 WMMA GEMM for gfx1250 (MI455X).
//   out[b,s,o] = sum_k x[b,s,k] * ((nibble(o,k) - 8) * scale[o]) + bias[o]
// Reformulated: acc = x_f16 @ Wq_f16^T (f32 accum), out = acc*scale + bias.
// M = B*S = 8192, K = 4096, N = 11008.
// ---------------------------------------------------------------------------

typedef _Float16 v8h  __attribute__((ext_vector_type(8)));
typedef _Float16 v16h __attribute__((ext_vector_type(16)));
typedef float    v8f  __attribute__((ext_vector_type(8)));

#define K_DIM   4096
#define KPACK   (K_DIM / 2)    // packed int32 columns (1 byte value per int32)
#define N_DIM   11008

#define TILE_M  128
#define TILE_N  128
#define TILE_K  64
#define LDS_STRIDE 88          // halves; 176B rows: 16B aligned, bank-conflict-spread

// Fragment loader for the ISA 16-bit 16x32 operand layout:
// lanes 0-15 (row = laneLow): halves K = kb..kb+7 and kb+16..kb+23 with kb = ks
// lanes 16-31 (row = laneLow): same with kb = ks+8
__device__ __forceinline__ v16h load_frag(const _Float16* __restrict__ tile,
                                          int row, int kb) {
  const _Float16* p = tile + row * LDS_STRIDE + kb;
  v8h lo = *(const v8h*)(p);        // 16B aligned ds_load_b128
  v8h hi = *(const v8h*)(p + 16);
  v16h r;
#pragma unroll
  for (int i = 0; i < 8; ++i) { r[i] = lo[i]; r[i + 8] = hi[i]; }
  return r;
}

// Convert 8 f32 -> 8 f16 and store as one b128 to LDS.
__device__ __forceinline__ void stage_x(_Float16* __restrict__ dst,
                                        const float4* __restrict__ xr) {
#pragma unroll
  for (int i = 0; i < 4; ++i) {
    float4 a = xr[2 * i], b = xr[2 * i + 1];
    v8h h = { (_Float16)a.x, (_Float16)a.y, (_Float16)a.z, (_Float16)a.w,
              (_Float16)b.x, (_Float16)b.y, (_Float16)b.z, (_Float16)b.w };
    *(v8h*)(dst + 8 * i) = h;
  }
}

// Unpack 4 int32 (each holding one byte = two nibbles) -> 8 f16 ints in [-8,7].
__device__ __forceinline__ void stage_w(_Float16* __restrict__ dst,
                                        const int4* __restrict__ wr) {
#pragma unroll
  for (int i = 0; i < 4; ++i) {
    int q0 = wr[i].x, q1 = wr[i].y, q2 = wr[i].z, q3 = wr[i].w;
    v8h h = { (_Float16)(short)(( q0        & 15) - 8),
              (_Float16)(short)(((q0 >>  4) & 15) - 8),
              (_Float16)(short)(( q1        & 15) - 8),
              (_Float16)(short)(((q1 >>  4) & 15) - 8),
              (_Float16)(short)(( q2        & 15) - 8),
              (_Float16)(short)(((q2 >>  4) & 15) - 8),
              (_Float16)(short)(( q3        & 15) - 8),
              (_Float16)(short)(((q3 >>  4) & 15) - 8) };
    *(v8h*)(dst + 8 * i) = h;
  }
}

__global__ __launch_bounds__(256)
void qlinear_wmma_f16_kernel(const float* __restrict__ x,
                             const int*   __restrict__ wpk,
                             const float* __restrict__ scale,
                             const float* __restrict__ bias,
                             float* __restrict__ out) {
  extern __shared__ _Float16 smem[];
  _Float16* As = smem;                                 // 2 x 128 x 88 halves
  _Float16* Bs = smem + 2 * TILE_M * LDS_STRIDE;       // 2 x 128 x 88 halves
  const int bufHalves = TILE_M * LDS_STRIDE;

  const int tid     = threadIdx.x;
  const int lane    = tid & 31;
  const int laneLow = lane & 15;
  const int laneHi8 = (lane >> 4) * 8;   // K sub-offset for A/B frags, M sub-offset for C/D
  const int wave    = tid >> 5;          // 8 waves
  const int wm      = wave >> 2;         // 0..1  -> 64 rows of M
  const int wn      = wave & 3;          // 0..3  -> 32 cols of N

  const int m0 = blockIdx.y * TILE_M;
  const int n0 = blockIdx.x * TILE_N;

  // Staging coords: 256 threads cover 128 rows x 2 column-segments.
  const int srow = tid >> 1;             // 0..127
  const int sseg = tid & 1;              // 0/1

  const float* xg = x   + (size_t)(m0 + srow) * K_DIM + sseg * 32;  // 32 f32 / thread / stage
  const int*   wg = wpk + (size_t)(n0 + srow) * KPACK + sseg * 16;  // 16 i32 / thread / stage

  _Float16* aDst = As + srow * LDS_STRIDE + sseg * 32;
  _Float16* bDst = Bs + srow * LDS_STRIDE + sseg * 32;

  v8f acc[4][2];
#pragma unroll
  for (int i = 0; i < 4; ++i)
#pragma unroll
    for (int j = 0; j < 2; ++j) {
      v8f z = {};
      acc[i][j] = z;
    }

  float4 xr[8];
  int4   wr[4];

  const int nk = K_DIM / TILE_K;   // 64 stages

  // Prologue: stage 0 -> buffer 0.
  {
    const float4* xp = (const float4*)xg;
#pragma unroll
    for (int i = 0; i < 8; ++i) xr[i] = xp[i];
    const int4* wp = (const int4*)wg;
#pragma unroll
    for (int i = 0; i < 4; ++i) wr[i] = wp[i];
    stage_x(aDst, xr);
    stage_w(bDst, wr);
  }
  __syncthreads();

  for (int kt = 0; kt < nk; ++kt) {
    const int  cur     = kt & 1;
    const bool hasNext = (kt + 1) < nk;

    // Issue next stage's global loads first so they overlap the WMMAs.
    if (hasNext) {
      const float4* xp = (const float4*)(xg + (size_t)(kt + 1) * TILE_K);
#pragma unroll
      for (int i = 0; i < 8; ++i) xr[i] = xp[i];
      const int4* wp = (const int4*)(wg + (size_t)(kt + 1) * (TILE_K / 2));
#pragma unroll
      for (int i = 0; i < 4; ++i) wr[i] = wp[i];
      if (kt + 2 < nk) {  // gfx1250 global_prefetch of the stage after next
        __builtin_prefetch(xg + (size_t)(kt + 2) * TILE_K, 0, 1);
        __builtin_prefetch(wg + (size_t)(kt + 2) * (TILE_K / 2), 0, 1);
      }
    }

    const _Float16* Ab = As + cur * bufHalves;
    const _Float16* Bb = Bs + cur * bufHalves;

#pragma unroll
    for (int ks = 0; ks < TILE_K; ks += 32) {
      v16h af[4], bf[2];
#pragma unroll
      for (int i = 0; i < 4; ++i)
        af[i] = load_frag(Ab, wm * 64 + i * 16 + laneLow, ks + laneHi8);
#pragma unroll
      for (int j = 0; j < 2; ++j)
        bf[j] = load_frag(Bb, wn * 32 + j * 16 + laneLow, ks + laneHi8);
#pragma unroll
      for (int i = 0; i < 4; ++i)
#pragma unroll
        for (int j = 0; j < 2; ++j)
          acc[i][j] = __builtin_amdgcn_wmma_f32_16x16x32_f16(
              false, af[i], false, bf[j], (short)0, acc[i][j], false, false);
    }

    if (hasNext) {
      const int nxt = cur ^ 1;
      stage_x(As + nxt * bufHalves + srow * LDS_STRIDE + sseg * 32, xr);
      stage_w(Bs + nxt * bufHalves + srow * LDS_STRIDE + sseg * 32, wr);
      __syncthreads();  // single barrier per stage: publishes nxt, fences cur readers
    }
  }

  // Epilogue: apply per-output-channel scale + bias on the f32 accumulators.
  // C/D layout: VGPR r -> M = r (lanes 0-15) or M = r+8 (lanes 16-31), N = lane%16.
#pragma unroll
  for (int j = 0; j < 2; ++j) {
    const int   n  = n0 + wn * 32 + j * 16 + laneLow;
    const float sc = scale[n];
    const float bi = bias[n];
#pragma unroll
    for (int i = 0; i < 4; ++i) {
      const int mb = m0 + wm * 64 + i * 16 + laneHi8;
      float* op = out + (size_t)mb * N_DIM + n;
#pragma unroll
      for (int r = 0; r < 8; ++r)
        op[(size_t)r * N_DIM] = acc[i][j][r] * sc + bi;
    }
  }
}

extern "C" void kernel_launch(void* const* d_in, const int* in_sizes, int n_in,
                              void* d_out, int out_size, void* d_ws, size_t ws_size,
                              hipStream_t stream) {
  (void)n_in; (void)out_size; (void)d_ws; (void)ws_size;
  const float* x     = (const float*)d_in[0];
  const int*   wpk   = (const int*)d_in[1];
  const float* scale = (const float*)d_in[2];
  const float* bias  = (const float*)d_in[3];
  float*       out   = (float*)d_out;

  const int M = in_sizes[0] / K_DIM;                     // 8192
  const size_t ldsBytes = 4u * TILE_M * LDS_STRIDE * sizeof(_Float16);  // 90112 B

  static_assert(4u * TILE_M * LDS_STRIDE * sizeof(_Float16) <= 160 * 1024, "LDS budget");

  hipFuncSetAttribute((const void*)qlinear_wmma_f16_kernel,
                      hipFuncAttributeMaxDynamicSharedMemorySize,
                      (int)ldsBytes);

  dim3 grid(N_DIM / TILE_N, M / TILE_M);  // (86, 64)
  qlinear_wmma_f16_kernel<<<grid, 256, ldsBytes, stream>>>(x, wpk, scale, bias, out);
}
